// WireframeParser_30992484008078
// MI455X (gfx1250) — matrix operand order repaired and checked by reading, not compiled, not run on hardware
//
#include <hip/hip_runtime.h>
#include <hip/hip_bf16.h>
#include <math.h>

typedef _Float16 half_t;
typedef __attribute__((ext_vector_type(16))) _Float16 v16h;
typedef __attribute__((ext_vector_type(8)))  _Float16 v8h;
typedef __attribute__((ext_vector_type(4)))  _Float16 v4h;
typedef __attribute__((ext_vector_type(8)))  float    v8f;
typedef __attribute__((ext_vector_type(4)))  float    v4f;

constexpr int H_    = 128;
constexpr int W_    = 128;
constexpr int HW_   = H_ * W_;
constexpr int TOPK_ = 300;
constexpr int L_    = 8192;
constexpr int DLOI_ = 128;

// ---------------------------------------------------------------------------
// Stage 1: jloc = softmax(out[5:7])[1] = sigmoid(o6-o5);  joff = sigmoid-0.5
// ---------------------------------------------------------------------------
__global__ __launch_bounds__(256) void prep_kernel(
    const float* __restrict__ out_map, float* __restrict__ a_map,
    float* __restrict__ joff)
{
    int i = blockIdx.x * 256 + threadIdx.x;
    if (i >= HW_) return;
    float o5 = out_map[5 * HW_ + i], o6 = out_map[6 * HW_ + i];
    a_map[i] = 1.0f / (1.0f + __expf(o5 - o6));
    float o7 = out_map[7 * HW_ + i], o8 = out_map[8 * HW_ + i];
    joff[i]       = 1.0f / (1.0f + __expf(-o7)) - 0.5f;
    joff[HW_ + i] = 1.0f / (1.0f + __expf(-o8)) - 0.5f;
}

// ---------------------------------------------------------------------------
// Stage 2: 3x3 NMS (SAME padding with -inf identity == skip OOB)
// ---------------------------------------------------------------------------
__global__ __launch_bounds__(256) void nms_kernel(
    const float* __restrict__ a_map, float* __restrict__ jflat)
{
    int i = blockIdx.x * 256 + threadIdx.x;
    if (i >= HW_) return;
    int y = i >> 7, x = i & 127;
    float a = a_map[i];
    float mx = -INFINITY;
    for (int dy = -1; dy <= 1; ++dy) {
        int yy = y + dy; if (yy < 0 || yy > H_ - 1) continue;
        for (int dx = -1; dx <= 1; ++dx) {
            int xx = x + dx; if (xx < 0 || xx > W_ - 1) continue;
            mx = fmaxf(mx, a_map[yy * W_ + xx]);
        }
    }
    jflat[i] = (a == mx) ? a : 0.0f;
}

// ---------------------------------------------------------------------------
// Stage 3: top-300 (single 1024-thread workgroup, 32 wave32 waves) + juncs
// ---------------------------------------------------------------------------
__global__ __launch_bounds__(1024) void topk_kernel(
    float* __restrict__ jflat, const float* __restrict__ joff,
    int* __restrict__ tkidx, float* __restrict__ tkval,
    float* __restrict__ juncs)
{
    __shared__ float swv[32];
    __shared__ int   swi[32];
    const int tid  = threadIdx.x;
    const int lane = tid & 31;
    const int wv   = tid >> 5;

    for (int iter = 0; iter < TOPK_; ++iter) {
        float best = -1e30f; int bidx = 0;
        for (int i = tid; i < HW_; i += 1024) {
            float v = jflat[i];
            if (v > best) { best = v; bidx = i; }
        }
        #pragma unroll
        for (int off = 16; off > 0; off >>= 1) {
            float ov = __shfl_down(best, off, 32);
            int   oi = __shfl_down(bidx, off, 32);
            if (ov > best || (ov == best && oi < bidx)) { best = ov; bidx = oi; }
        }
        if (lane == 0) { swv[wv] = best; swi[wv] = bidx; }
        __syncthreads();
        if (wv == 0) {
            best = swv[lane]; bidx = swi[lane];
            #pragma unroll
            for (int off = 16; off > 0; off >>= 1) {
                float ov = __shfl_down(best, off, 32);
                int   oi = __shfl_down(bidx, off, 32);
                if (ov > best || (ov == best && oi < bidx)) { best = ov; bidx = oi; }
            }
            if (lane == 0) {
                tkidx[iter] = bidx;
                tkval[iter] = best;
                jflat[bidx] = -1e30f;     // remove from further rounds
                __threadfence_block();
            }
        }
        __syncthreads();
    }
    __threadfence_block();
    __syncthreads();
    for (int t = tid; t < TOPK_; t += 1024) {
        int id  = tkidx[t];
        float x = (float)(id & (W_ - 1)) + joff[id]        + 0.5f;
        float y = (float)(id >> 7)       + joff[HW_ + id]  + 0.5f;
        juncs[2 * t + 0] = x;
        juncs[2 * t + 1] = y;
    }
}

// ---------------------------------------------------------------------------
// Stage 4: per-line nearest-junction matching (juncs staged in LDS)
// ---------------------------------------------------------------------------
__global__ __launch_bounds__(256) void match_kernel(
    const float* __restrict__ lines_raw, const float* __restrict__ juncs,
    float* __restrict__ lines_adj, int* __restrict__ iskeep)
{
    __shared__ float sj[TOPK_ * 2];
    for (int t = threadIdx.x; t < TOPK_ * 2; t += 256) sj[t] = juncs[t];
    __syncthreads();

    int l = blockIdx.x * 256 + threadIdx.x;
    if (l >= L_) return;
    float x1 = lines_raw[l * 4 + 0] * (float)W_;
    float y1 = lines_raw[l * 4 + 1] * (float)H_;
    float x2 = lines_raw[l * 4 + 2] * (float)W_;
    float y2 = lines_raw[l * 4 + 3] * (float)H_;
    float bd1 = 1e30f, bd2 = 1e30f; int i1 = 0, i2 = 0;
    for (int j = 0; j < TOPK_; ++j) {
        float jx = sj[2 * j], jy = sj[2 * j + 1];
        float ax = x1 - jx, ay = y1 - jy;
        float bx = x2 - jx, by = y2 - jy;
        float d1 = ax * ax + ay * ay;
        float d2 = bx * bx + by * by;
        if (d1 < bd1) { bd1 = d1; i1 = j; }
        if (d2 < bd2) { bd2 = d2; i2 = j; }
    }
    int imin = min(i1, i2), imax = max(i1, i2);
    iskeep[l] = (imin < imax) ? 1 : 0;
    lines_adj[l * 4 + 0] = sj[2 * imin];
    lines_adj[l * 4 + 1] = sj[2 * imin + 1];
    lines_adj[l * 4 + 2] = sj[2 * imax];
    lines_adj[l * 4 + 3] = sj[2 * imax + 1];
}

// ---------------------------------------------------------------------------
// Stage 5a: features (256 x 16384, f32) -> featA (16384 x 256, f16)
//   LDS-tiled transpose so both global read and write are coalesced.
// ---------------------------------------------------------------------------
__global__ __launch_bounds__(256) void transpose_kernel(
    const float* __restrict__ feats, half_t* __restrict__ featA)
{
    __shared__ float tile[64][65];
    const int m0 = blockIdx.x * 64;   // position
    const int k0 = blockIdx.y * 64;   // channel
    #pragma unroll
    for (int it = 0; it < 16; ++it) {
        int idx = it * 256 + threadIdx.x;
        int k = idx >> 6, m = idx & 63;
        tile[k][m] = feats[(size_t)(k0 + k) * HW_ + (m0 + m)];
    }
    __syncthreads();
    #pragma unroll
    for (int it = 0; it < 16; ++it) {
        int idx = it * 256 + threadIdx.x;
        int m = idx >> 6, k = idx & 63;
        featA[(size_t)(m0 + m) * 256 + (k0 + k)] = (half_t)tile[k][m];
    }
}

// ---------------------------------------------------------------------------
// Stage 5b: one-time weight conversion f32 -> f16 (vectorized copy)
// ---------------------------------------------------------------------------
__global__ __launch_bounds__(256) void f32to16_kernel(
    const float* __restrict__ src, half_t* __restrict__ dst, int n4)
{
    int i = blockIdx.x * 256 + threadIdx.x;
    if (i >= n4) return;
    v4f v = *(const v4f*)(src + (size_t)i * 4);
    v4h h;
    #pragma unroll
    for (int e = 0; e < 4; ++e) h[e] = (half_t)v[e];
    *(v4h*)(dst + (size_t)i * 4) = h;
}

// ---------------------------------------------------------------------------
// WMMA GEMM:  C(MxN) = act( A(MxK,f16) * W(NxK,f16)^T + bias )
//   - 256 threads = 8 wave32 waves; block tile 128x128; wave tile 32x64
//   - 8 WMMAs per wave per K-chunk, ONE barrier per chunk
//   - double-buffered LDS + register prefetch: next chunk's global loads
//     issue before this chunk's WMMAs (latency hidden under matrix ops)
//   - As row-major / Bs n-major, stride 40 halves: every fragment is two
//     contiguous v8h runs -> ds_load_b128, bank-conflict-free (20n mod 64)
// ---------------------------------------------------------------------------
#define BM 128
#define BN 128
#define KC 32
#define ASTR 40
#define KSTR 40
#define ABUF (BM * ASTR)
#define BBUF (BN * KSTR)

template <typename OUTT, bool RELU>
__global__ __launch_bounds__(256) void gemm_wmma(
    const half_t* __restrict__ A, const half_t* __restrict__ Wh,
    const float* __restrict__ bias, OUTT* __restrict__ C,
    int M, int N, int K)
{
    __shared__ __align__(16) half_t As[2 * ABUF];
    __shared__ __align__(16) half_t Bs[2 * BBUF];

    const int tid  = threadIdx.x;
    const int lane = tid & 31;
    const int wave = tid >> 5;       // 0..7
    const int wm   = wave & 3;       // M quadrant (4 x 32 rows)
    const int wn   = wave >> 2;      // N half (2 x 64 cols)
    const int m0   = blockIdx.x * BM;
    const int n0   = blockIdx.y * BN;

    // staging coordinates: 2 threads per row, 16 halves (b128 pair) each
    const int sr = tid >> 1;             // 0..127
    const int sk = (tid & 1) * 16;       // 0 / 16
    const half_t* Ag = &A [(size_t)(m0 + sr) * K + sk];
    const half_t* Bg = &Wh[(size_t)(n0 + sr) * K + sk];

    v8f acc[2][4];
    #pragma unroll
    for (int i = 0; i < 2; ++i)
        #pragma unroll
        for (int j = 0; j < 4; ++j) acc[i][j] = {};

    // ---- prologue: stage chunk 0 into buffer 0 ----
    {
        v8h a0 = *(const v8h*)Ag;
        v8h a1 = *(const v8h*)(Ag + 8);
        v8h b0 = *(const v8h*)Bg;
        v8h b1 = *(const v8h*)(Bg + 8);
        *(v8h*)&As[sr * ASTR + sk]     = a0;
        *(v8h*)&As[sr * ASTR + sk + 8] = a1;
        *(v8h*)&Bs[sr * KSTR + sk]     = b0;
        *(v8h*)&Bs[sr * KSTR + sk + 8] = b1;
    }
    __syncthreads();

    int cur = 0;
    for (int k0 = 0; k0 < K; k0 += KC) {
        const bool has_next = (k0 + KC) < K;

        // ---- issue next chunk's global loads early (hide under WMMA) ----
        v8h na0, na1, nb0, nb1;
        if (has_next) {
            const half_t* agn = Ag + k0 + KC;
            const half_t* bgn = Bg + k0 + KC;
            na0 = *(const v8h*)agn;
            na1 = *(const v8h*)(agn + 8);
            nb0 = *(const v8h*)bgn;
            nb1 = *(const v8h*)(bgn + 8);
            if (k0 + 2 * KC < K)   // CDNA5 global_prefetch_b8, chunk k+2
                __builtin_prefetch(Bg + k0 + 2 * KC, 0, 1);
        }

        // ---- fragments from current buffer: contiguous ds_load_b128 ----
        const half_t* Asc = &As[cur * ABUF];
        const half_t* Bsc = &Bs[cur * BBUF];
        const int r16 = lane & 15;
        const int kg  = lane >> 4;
        v16h afr[2], bfr[4];
        #pragma unroll
        for (int mi = 0; mi < 2; ++mi) {
            const half_t* ap = &Asc[(wm * 32 + mi * 16 + r16) * ASTR + kg * 8];
            v8h lo = *(const v8h*)ap;          // k = kg*8 + 0..7
            v8h hi = *(const v8h*)(ap + 16);   // k = 16 + kg*8 + 0..7
            afr[mi] = __builtin_shufflevector(lo, hi,
                0, 1, 2, 3, 4, 5, 6, 7, 8, 9, 10, 11, 12, 13, 14, 15);
        }
        #pragma unroll
        for (int ni = 0; ni < 4; ++ni) {
            const half_t* bp = &Bsc[(wn * 64 + ni * 16 + r16) * KSTR + kg * 8];
            v8h lo = *(const v8h*)bp;          // k = kg*8 + 0..7
            v8h hi = *(const v8h*)(bp + 16);   // k = 16 + kg*8 + 0..7
            bfr[ni] = __builtin_shufflevector(lo, hi,
                0, 1, 2, 3, 4, 5, 6, 7, 8, 9, 10, 11, 12, 13, 14, 15);
        }
        #pragma unroll
        for (int mi = 0; mi < 2; ++mi)
            #pragma unroll
            for (int ni = 0; ni < 4; ++ni)
                acc[mi][ni] = __builtin_amdgcn_wmma_f32_16x16x32_f16(
                    false, afr[mi], false, bfr[ni],
                    (short)0, acc[mi][ni], false, false);

        // ---- store next chunk into the other buffer ----
        if (has_next) {
            half_t* Asn = &As[(cur ^ 1) * ABUF];
            half_t* Bsn = &Bs[(cur ^ 1) * BBUF];
            *(v8h*)&Asn[sr * ASTR + sk]     = na0;
            *(v8h*)&Asn[sr * ASTR + sk + 8] = na1;
            *(v8h*)&Bsn[sr * KSTR + sk]     = nb0;
            *(v8h*)&Bsn[sr * KSTR + sk + 8] = nb1;
        }
        __syncthreads();
        cur ^= 1;
    }

    // ---- epilogue: bias + activation + store (lane>>4 selects M half) ----
    const int nlane = lane & 15;
    const int mg    = lane >> 4;
    #pragma unroll
    for (int mi = 0; mi < 2; ++mi)
        #pragma unroll
        for (int ni = 0; ni < 4; ++ni) {
            int ncol = n0 + wn * 64 + ni * 16 + nlane;
            float bv = bias[ncol];
            #pragma unroll
            for (int v = 0; v < 8; ++v) {
                int mrow2 = m0 + wm * 32 + mi * 16 + mg * 8 + v;
                float r = acc[mi][ni][v] + bv;
                if (RELU) r = fmaxf(r, 0.0f);
                C[(size_t)mrow2 * N + ncol] = (OUTT)r;
            }
        }
}

// ---------------------------------------------------------------------------
// Stage 6: bilinear line sampling + max-pool(4) -> feat_line (L x 1024) f16
//   loi stored position-major (16384 x 128) so loads coalesce over channel c
// ---------------------------------------------------------------------------
__global__ __launch_bounds__(128) void sample_kernel(
    const float* __restrict__ lines_adj, const float* __restrict__ loi_t,
    half_t* __restrict__ feat)
{
    int l = blockIdx.x;
    int c = threadIdx.x;                       // channel 0..127
    float ux = lines_adj[l * 4 + 0], uy = lines_adj[l * 4 + 1];
    float vx = lines_adj[l * 4 + 2], vy = lines_adj[l * 4 + 3];
    float mx[8];
    #pragma unroll
    for (int g = 0; g < 8; ++g) mx[g] = -1e30f;

    for (int p = 0; p < 32; ++p) {
        float t  = (float)p * (1.0f / 31.0f);
        float px = ux * t + vx * (1.0f - t) - 0.5f;
        float py = uy * t + vy * (1.0f - t) - 0.5f;
        float px0 = fminf(fmaxf(floorf(px), 0.0f), (float)(W_ - 1));
        float py0 = fminf(fmaxf(floorf(py), 0.0f), (float)(H_ - 1));
        float px1 = fminf(px0 + 1.0f, (float)(W_ - 1));
        float py1 = fminf(py0 + 1.0f, (float)(H_ - 1));
        int x0 = (int)px0, y0 = (int)py0, x1 = (int)px1, y1 = (int)py1;
        float v00 = loi_t[(size_t)(y0 * W_ + x0) * DLOI_ + c];
        float v10 = loi_t[(size_t)(y1 * W_ + x0) * DLOI_ + c];
        float v01 = loi_t[(size_t)(y0 * W_ + x1) * DLOI_ + c];
        float v11 = loi_t[(size_t)(y1 * W_ + x1) * DLOI_ + c];
        float val = v00 * (py1 - py) * (px1 - px)
                  + v10 * (py - py0) * (px1 - px)
                  + v01 * (py1 - py) * (px - px0)
                  + v11 * (py - py0) * (px - px0);
        mx[p >> 2] = fmaxf(mx[p >> 2], val);
    }
    #pragma unroll
    for (int g = 0; g < 8; ++g)
        feat[(size_t)l * 1024 + c * 8 + g] = (half_t)mx[g];
}

// ---------------------------------------------------------------------------
// Stage 9: final dot + sigmoid + iskeep mask (one wave32 per line)
// ---------------------------------------------------------------------------
__global__ __launch_bounds__(256) void head_kernel(
    const half_t* __restrict__ h2, const float* __restrict__ w3,
    const float* __restrict__ b3, const int* __restrict__ iskeep,
    float* __restrict__ outp)
{
    int lane = threadIdx.x & 31;
    int wv   = threadIdx.x >> 5;
    int l    = blockIdx.x * 8 + wv;
    if (l >= L_) return;
    const half_t* hp = h2 + (size_t)l * 1024;
    float s = 0.0f;
    for (int k = lane; k < 1024; k += 32)
        s += (float)hp[k] * w3[k];
    #pragma unroll
    for (int off = 16; off > 0; off >>= 1)
        s += __shfl_down(s, off, 32);
    if (lane == 0) {
        float logit = s + b3[0];
        float sg = 1.0f / (1.0f + __expf(-logit));
        outp[l] = iskeep[l] ? sg : 0.0f;
    }
}

// ---------------------------------------------------------------------------
extern "C" void kernel_launch(void* const* d_in, const int* in_sizes, int n_in,
                              void* d_out, int out_size, void* d_ws, size_t ws_size,
                              hipStream_t stream)
{
    (void)in_sizes; (void)n_in; (void)out_size; (void)ws_size;
    const float* output_map = (const float*)d_in[0];
    const float* features   = (const float*)d_in[1];
    const float* lines_raw  = (const float*)d_in[2];
    const float* fc1_w      = (const float*)d_in[3];
    const float* fc1_b      = (const float*)d_in[4];
    const float* w1         = (const float*)d_in[5];
    const float* b1         = (const float*)d_in[6];
    const float* w2         = (const float*)d_in[7];
    const float* b2         = (const float*)d_in[8];
    const float* w3         = (const float*)d_in[9];
    const float* b3         = (const float*)d_in[10];

    char* p = (char*)d_ws;
    auto alloc = [&](size_t bytes) -> char* {
        char* r = p;
        p += (bytes + 255) & ~(size_t)255;
        return r;
    };
    float*  a_map  = (float*)alloc((size_t)HW_ * 4);
    float*  joff   = (float*)alloc((size_t)2 * HW_ * 4);
    float*  jflat  = (float*)alloc((size_t)HW_ * 4);
    int*    tkidx  = (int*)  alloc((size_t)TOPK_ * 4);
    float*  tkval  = (float*)alloc((size_t)TOPK_ * 4);
    float*  juncs  = (float*)alloc((size_t)TOPK_ * 2 * 4);
    float*  ladj   = (float*)alloc((size_t)L_ * 4 * 4);
    int*    keep   = (int*)  alloc((size_t)L_ * 4);
    half_t* featA  = (half_t*)alloc((size_t)HW_ * 256 * 2);    //  8 MB
    half_t* fc1_wh = (half_t*)alloc((size_t)DLOI_ * 256 * 2);  // 64 KB
    half_t* w1h    = (half_t*)alloc((size_t)1024 * 1024 * 2);  //  2 MB
    half_t* w2h    = (half_t*)alloc((size_t)1024 * 1024 * 2);  //  2 MB
    float*  loi_t  = (float*)alloc((size_t)HW_ * DLOI_ * 4);   //  8 MB
    half_t* feat   = (half_t*)alloc((size_t)L_ * 1024 * 2);    // 16 MB
    half_t* h1     = (half_t*)alloc((size_t)L_ * 1024 * 2);    // 16 MB
    half_t* h2     = (half_t*)alloc((size_t)L_ * 1024 * 2);    // 16 MB

    prep_kernel <<<HW_ / 256, 256, 0, stream>>>(output_map, a_map, joff);
    nms_kernel  <<<HW_ / 256, 256, 0, stream>>>(a_map, jflat);
    topk_kernel <<<1, 1024, 0, stream>>>(jflat, joff, tkidx, tkval, juncs);
    match_kernel<<<L_ / 256, 256, 0, stream>>>(lines_raw, juncs, ladj, keep);

    // features -> f16 position-major A matrix; weights -> f16 (one-time)
    dim3 g_tr(HW_ / 64, 256 / 64);
    transpose_kernel<<<g_tr, 256, 0, stream>>>(features, featA);
    f32to16_kernel<<<(DLOI_ * 256 / 4 + 255) / 256, 256, 0, stream>>>(
        fc1_w, fc1_wh, DLOI_ * 256 / 4);
    f32to16_kernel<<<(1024 * 1024 / 4 + 255) / 256, 256, 0, stream>>>(
        w1, w1h, 1024 * 1024 / 4);
    f32to16_kernel<<<(1024 * 1024 / 4 + 255) / 256, 256, 0, stream>>>(
        w2, w2h, 1024 * 1024 / 4);

    // loi^T (16384 x 128) = featA (16384x256,f16) * fc1_w^T + fc1_b
    dim3 g_loi(HW_ / BM, DLOI_ / BN);
    gemm_wmma<float, false><<<g_loi, 256, 0, stream>>>(
        featA, fc1_wh, fc1_b, loi_t, HW_, DLOI_, 256);

    sample_kernel<<<L_, 128, 0, stream>>>(ladj, loi_t, feat);

    // MLP: two 8192x1024x1024 WMMA GEMMs with fused bias+ReLU, f16 activations
    dim3 g_mlp(L_ / BM, 1024 / BN);
    gemm_wmma<half_t, true><<<g_mlp, 256, 0, stream>>>(
        feat, w1h, b1, h1, L_, 1024, 1024);
    gemm_wmma<half_t, true><<<g_mlp, 256, 0, stream>>>(
        h1, w2h, b2, h2, L_, 1024, 1024);

    head_kernel<<<L_ / 8, 256, 0, stream>>>(h2, w3, b3, keep, (float*)d_out);
}
